// L2AC_25709674233906
// MI455X (gfx1250) — matrix operand order repaired and checked by reading, not compiled, not run on hardware
//
#include <hip/hip_runtime.h>
#include <math.h>

typedef __attribute__((ext_vector_type(16))) __bf16 v16bf;
typedef __attribute__((ext_vector_type(8)))  __bf16 v8bf;
typedef __attribute__((ext_vector_type(8)))  float   v8f;
typedef unsigned int u32x4 __attribute__((ext_vector_type(4)));
typedef int          i32x8 __attribute__((ext_vector_type(8)));
typedef int          i32x4 __attribute__((ext_vector_type(4)));

constexpr int DIM  = 2048;   // D
constexpr int SEQ  = 10240;  // S
constexpr int KDIM = 4096;   // 2*D
constexpr int MT   = 128;    // M tile
constexpr int NT   = 256;    // N tile
constexpr int KT   = 32;     // K chunk (bf16 WMMA K)
constexpr int APAD = 40;     // padded LDS row stride (elems) for A
constexpr int BPAD = 40;     // padded LDS row stride (elems) for B (64B row + 16B pad)
constexpr int NPART = 16;    // deterministic partial slots (8 N-blocks * 2 N-waves)
constexpr int NKC  = KDIM / KT;  // 128 K chunks

#if __has_builtin(__builtin_amdgcn_tensor_load_to_lds) && __has_builtin(__builtin_amdgcn_s_wait_tensorcnt)
#define HAVE_TDM 1
#else
#define HAVE_TDM 0
#endif

__device__ __forceinline__ float sigmoid_fast(float x) {
    return 1.0f / (1.0f + __expf(-x));
}
__device__ __forceinline__ float tanh_fast(float x) {
    float e = __expf(2.0f * x);
    return 1.0f - 2.0f / (e + 1.0f);   // stable at +/- inf
}

#if HAVE_TDM
// 32-bit LDS byte offset of a __shared__ object (ptrtoint of an AS(3) pointer)
__device__ __forceinline__ unsigned lds_off(const void* p) {
    return (unsigned)(size_t)(const __attribute__((address_space(3))) void*)p;
}

// Issue a TDM 2D tile load: tile_dim0=KT bf16 (64B rows), tile_dim1=NT rows,
// row stride KDIM elems; LDS padded 16B per 64B (pad_interval=3 -> 16 dwords,
// pad_amount=3 -> 4 dwords) to match BPAD=40-element rows.
__device__ __forceinline__ void tdm_load_b(const __bf16* gsrc, unsigned ldst,
                                           int rem_k, int rem_n) {
    u32x4 g0;
    unsigned long long ga = (unsigned long long)gsrc;
    g0[0] = 1u;                                  // count=1, user descriptor
    g0[1] = ldst;                                // lds_addr
    g0[2] = (unsigned)ga;                        // global_addr[31:0]
    g0[3] = (unsigned)(ga >> 32) | (2u << 30);   // global_addr[56:32] | type=2
    i32x8 g1;
    g1[0] = (1 << 16) | (1 << 20) | (3 << 22) | (3 << 25); // data_size=2B, pad on
    g1[1] = (int)(((unsigned)rem_k & 0xFFFFu) << 16);      // tensor_dim0 lo
    g1[2] = (int)((((unsigned)rem_k) >> 16) |
                  (((unsigned)rem_n & 0xFFFFu) << 16));    // dim0 hi | dim1 lo
    g1[3] = (int)((((unsigned)rem_n) >> 16) |
                  ((unsigned)KT << 16));                   // dim1 hi | tile_dim0
    g1[4] = NT;                                  // tile_dim1=256, tile_dim2=0
    g1[5] = KDIM;                                // tensor_dim0_stride lo (elems)
    g1[6] = 0;                                   // stride hi | dim1_stride lo
    g1[7] = 0;
    i32x4 gz4 = {0, 0, 0, 0};
    i32x8 gz8 = {0, 0, 0, 0, 0, 0, 0, 0};
    __builtin_amdgcn_tensor_load_to_lds(g0, g1, gz4, gz4, gz8, 0);
    asm volatile("" ::: "memory");
}
#endif

// ---------------------------------------------------------------------------
// Prep: W1t[n][k] (bf16) = W1[k][n] (f32) — LDS-tiled transpose + convert.
// grid (DIM/32, KDIM/32), block 256
// ---------------------------------------------------------------------------
__global__ __launch_bounds__(256)
void k_prep(const float* __restrict__ W1, __bf16* __restrict__ W1t) {
    __shared__ float tile[32][33];
    const int n0 = blockIdx.x * 32;
    const int k0 = blockIdx.y * 32;
    const int tx = threadIdx.x & 31;
    const int ty = threadIdx.x >> 5;        // 8 row-groups
    #pragma unroll
    for (int r = ty; r < 32; r += 8)
        tile[r][tx] = W1[(size_t)(k0 + r) * DIM + n0 + tx];
    __syncthreads();
    #pragma unroll
    for (int r = ty; r < 32; r += 8)
        W1t[(size_t)(n0 + r) * KDIM + k0 + tx] = (__bf16)tile[tx][r];
}

// ===========================================================================
// Shared GEMM compute machinery (both variants)
// pre[s] partial = sum_d relu( sum_f sim[s,f]*W1[f,d] + b1[d] ) * W2[d]
// ===========================================================================
__device__ __forceinline__ v16bf load_frag16(const __bf16* p, int gap) {
    v8bf lo = *(const v8bf*)p;
    v8bf hi = *(const v8bf*)(p + gap);
    return __builtin_shufflevector(lo, hi,
        0, 1, 2, 3, 4, 5, 6, 7, 8, 9, 10, 11, 12, 13, 14, 15);
}

__device__ __forceinline__ void gemm_compute_chunk(
        const __bf16* Alds, const __bf16* Bbuf,
        int wm, int wn, int llo, int lhi, v8f acc[2][8]) {
    // A fragments: lane l<16 row M=l (K 0..7 / 16..23), lane>=16 (K 8..15 / 24..31)
    v16bf afrag[2];
    #pragma unroll
    for (int sub = 0; sub < 2; ++sub)
        afrag[sub] = load_frag16(
            &Alds[(wm * 32 + sub * 16 + llo) * APAD + lhi * 8], 16);

    // Process N sub-tiles in groups of 4: prefetch 4 B fragments into
    // distinct registers, then issue 8 back-to-back WMMAs (XDL-friendly).
    #pragma unroll
    for (int grp = 0; grp < 2; ++grp) {
        v16bf bfrag[4];
        #pragma unroll
        for (int q = 0; q < 4; ++q) {
            int nt = grp * 4 + q;
            bfrag[q] = load_frag16(
                &Bbuf[(wn * 128 + nt * 16 + llo) * BPAD + lhi * 8], 16);
        }
        #pragma unroll
        for (int q = 0; q < 4; ++q) {
            int nt = grp * 4 + q;
            #pragma unroll
            for (int sub = 0; sub < 2; ++sub) {
                acc[sub][nt] = __builtin_amdgcn_wmma_f32_16x16x32_bf16(
                    false, afrag[sub], false, bfrag[q],
                    (short)0, acc[sub][nt], false, false);
            }
        }
    }
}

__device__ __forceinline__ void gemm_epilogue(
        const float* b1, const float* W2, float* xpart,
        int s0, int n0, int wm, int wn, int llo, int lhi,
        int part, v8f acc[2][8]) {
    float b1v[8], w2v[8];
    #pragma unroll
    for (int nt = 0; nt < 8; ++nt) {
        int col = n0 + wn * 128 + nt * 16 + llo;
        b1v[nt] = b1[col];
        w2v[nt] = W2[col];
    }
    #pragma unroll
    for (int sub = 0; sub < 2; ++sub) {
        #pragma unroll
        for (int j = 0; j < 8; ++j) {
            float p = 0.0f;
            #pragma unroll
            for (int nt = 0; nt < 8; ++nt) {
                float pre = acc[sub][nt][j] + b1v[nt];
                p += fmaxf(pre, 0.0f) * w2v[nt];
            }
            p += __shfl_xor(p, 1);
            p += __shfl_xor(p, 2);
            p += __shfl_xor(p, 4);
            p += __shfl_xor(p, 8);
            if (llo == 0) {
                int row = s0 + wm * 32 + sub * 16 + j + lhi * 8;
                xpart[(size_t)part * SEQ + row] = p;
            }
        }
    }
}

// ---------------------------------------------------------------------------
// GEMM variant A: TDM-staged B tiles from pre-converted W1t (bf16, n-major).
// Double-buffered: chunk k+1 DMA overlaps chunk k WMMA.
// ---------------------------------------------------------------------------
__global__ __launch_bounds__(256)
void k_gemm_tdm(const float* __restrict__ x0,
                const float* __restrict__ x1,
                const __bf16* __restrict__ W1t,
                const float* __restrict__ b1,
                const float* __restrict__ W2,
                float* __restrict__ xpart) {
    __shared__ __align__(16) __bf16 Alds[MT * APAD];        // 10240 B
    __shared__ __align__(16) __bf16 Blds[2][NT * BPAD];     // 2 x 20480 B
    __shared__ float x0s[DIM];                              // 8192 B

    const int t  = threadIdx.x;
    const int s0 = blockIdx.x * MT;
    const int n0 = blockIdx.y * NT;

    for (int i = t; i < DIM; i += 256) x0s[i] = x0[i];

    const int w    = t >> 5;
    const int lane = t & 31;
    const int wm   = w & 3;
    const int wn   = w >> 2;
    const int llo  = lane & 15;
    const int lhi  = lane >> 4;

    v8f zero = {};
    v8f acc[2][8];
    #pragma unroll
    for (int sub = 0; sub < 2; ++sub)
        #pragma unroll
        for (int nt = 0; nt < 8; ++nt) acc[sub][nt] = zero;

    const int ar = t >> 1;
    const int ak = (t & 1) * 16;

    const __bf16* Wtile = W1t + (size_t)n0 * KDIM;   // tile origin column block

#if HAVE_TDM
    if (w == 0) {
        tdm_load_b(Wtile, lds_off(&Blds[0][0]), KDIM, DIM - n0);
    }
#else
    {   // fallback: contiguous bf16 vector copy of chunk 0 (row n = t)
        const v8bf* src = (const v8bf*)(W1t + (size_t)(n0 + t) * KDIM);
        v8bf* dst = (v8bf*)&Blds[0][t * BPAD];
        #pragma unroll
        for (int i = 0; i < 4; ++i) dst[i] = src[i];
    }
#endif

    for (int kc = 0; kc < NKC; ++kc) {
        const int  k0   = kc * KT;
        const bool h2   = (k0 >= DIM);
        const int  dim0 = h2 ? (k0 - DIM) : k0;
        const int  cur  = kc & 1;

        __syncthreads();   // previous compute finished reading Alds / Blds[1-cur]
        // ---- stage A tile (on-the-fly sim, bf16) --------------------------
        {
            const float* xr  = x1 + (size_t)(s0 + ar) * DIM + dim0 + ak;
            const float* x0p = x0s + dim0 + ak;
            __builtin_prefetch(xr + KT, 0, 1);
            #pragma unroll
            for (int i = 0; i < 16; ++i) {
                float xv = xr[i];
                float ov = x0p[i];
                float sv = h2 ? (ov + xv) : fabsf(ov - xv);
                Alds[ar * APAD + ak + i] = (__bf16)sv;
            }
        }
#if HAVE_TDM
        if (w == 0) __builtin_amdgcn_s_wait_tensorcnt(0);   // Blds[cur] ready
#endif
        __syncthreads();

        // ---- kick DMA for next chunk into the other buffer ----------------
        if (kc + 1 < NKC) {
#if HAVE_TDM
            if (w == 0) {
                tdm_load_b(Wtile + (size_t)(k0 + KT),
                           lds_off(&Blds[1 - cur][0]),
                           KDIM - (k0 + KT), DIM - n0);
            }
#else
            const v8bf* src = (const v8bf*)(W1t + (size_t)(n0 + t) * KDIM + k0 + KT);
            v8bf* dst = (v8bf*)&Blds[1 - cur][t * BPAD];
            #pragma unroll
            for (int i = 0; i < 4; ++i) dst[i] = src[i];
#endif
        }

        // ---- WMMA on current buffers --------------------------------------
        gemm_compute_chunk(Alds, &Blds[cur][0], wm, wn, llo, lhi, acc);
    }

    gemm_epilogue(b1, W2, xpart, s0, n0, wm, wn, llo, lhi,
                  blockIdx.y * 2 + wn, acc);
}

// ---------------------------------------------------------------------------
// GEMM variant B (small-ws fallback): direct f32 W1 staging.
// ---------------------------------------------------------------------------
__global__ __launch_bounds__(256)
void k_gemm_direct(const float* __restrict__ x0,
                   const float* __restrict__ x1,
                   const float* __restrict__ W1,
                   const float* __restrict__ b1,
                   const float* __restrict__ W2,
                   float* __restrict__ xpart) {
    __shared__ __align__(16) __bf16 Alds[MT * APAD];
    __shared__ __align__(16) __bf16 Blds[NT * BPAD];
    __shared__ float x0s[DIM];

    const int t  = threadIdx.x;
    const int s0 = blockIdx.x * MT;
    const int n0 = blockIdx.y * NT;

    for (int i = t; i < DIM; i += 256) x0s[i] = x0[i];

    const int w    = t >> 5;
    const int lane = t & 31;
    const int wm   = w & 3;
    const int wn   = w >> 2;
    const int llo  = lane & 15;
    const int lhi  = lane >> 4;

    v8f zero = {};
    v8f acc[2][8];
    #pragma unroll
    for (int sub = 0; sub < 2; ++sub)
        #pragma unroll
        for (int nt = 0; nt < 8; ++nt) acc[sub][nt] = zero;

    const int ar = t >> 1;
    const int ak = (t & 1) * 16;

    for (int kc = 0; kc < NKC; ++kc) {
        const int  k0   = kc * KT;
        const bool h2   = (k0 >= DIM);
        const int  dim0 = h2 ? (k0 - DIM) : k0;

        __syncthreads();
        {
            const float* xr  = x1 + (size_t)(s0 + ar) * DIM + dim0 + ak;
            const float* x0p = x0s + dim0 + ak;
            #pragma unroll
            for (int i = 0; i < 16; ++i) {
                float xv = xr[i];
                float ov = x0p[i];
                float sv = h2 ? (ov + xv) : fabsf(ov - xv);
                Alds[ar * APAD + ak + i] = (__bf16)sv;
            }
        }
        {
            #pragma unroll
            for (int i = 0; i < KT; ++i) {
                float wv = W1[(size_t)(k0 + i) * DIM + n0 + t];
                Blds[t * BPAD + i] = (__bf16)wv;
            }
            if (kc + 1 < NKC)
                __builtin_prefetch(W1 + (size_t)(k0 + KT) * DIM + n0 + t, 0, 1);
        }
        __syncthreads();

        gemm_compute_chunk(Alds, Blds, wm, wn, llo, lhi, acc);
    }

    gemm_epilogue(b1, W2, xpart, s0, n0, wm, wn, llo, lhi,
                  blockIdx.y * 2 + wn, acc);
}

// ---------------------------------------------------------------------------
// LSTM: sum partials -> sigmoid -> bidirectional scalar scan (lanes 0/1)
// ---------------------------------------------------------------------------
__global__ __launch_bounds__(64)
void k_lstm(const float* __restrict__ xpart,
            const float* __restrict__ b2,
            const float* __restrict__ Wih_f, const float* __restrict__ Whh_f,
            const float* __restrict__ bih_f, const float* __restrict__ bhh_f,
            const float* __restrict__ Wih_b, const float* __restrict__ Whh_b,
            const float* __restrict__ bih_b, const float* __restrict__ bhh_b,
            float* __restrict__ flat) {
    __shared__ float xs[SEQ];  // 40 KB
    const int t = threadIdx.x;
    const float bb2 = b2[0];
    for (int i = t; i < SEQ; i += 64) {
        float acc = bb2;
        #pragma unroll
        for (int q = 0; q < NPART; ++q) acc += xpart[(size_t)q * SEQ + i];
        xs[i] = sigmoid_fast(acc);
    }
    __syncthreads();

    if (t < 2) {
        const float* Wih = (t == 0) ? Wih_f : Wih_b;
        const float* Whh = (t == 0) ? Whh_f : Whh_b;
        const float* bih = (t == 0) ? bih_f : bih_b;
        const float* bhh = (t == 0) ? bhh_f : bhh_b;
        const float wi0 = Wih[0], wi1 = Wih[1], wi2 = Wih[2], wi3 = Wih[3];
        const float wh0 = Whh[0], wh1 = Whh[1], wh2 = Whh[2], wh3 = Whh[3];
        const float g0b = bih[0] + bhh[0], g1b = bih[1] + bhh[1];
        const float g2b = bih[2] + bhh[2], g3b = bih[3] + bhh[3];
        float h = 0.0f, c = 0.0f;
        for (int j = 0; j < SEQ; ++j) {
            const int s = (t == 0) ? j : (SEQ - 1 - j);
            const float x = xs[s];
            float g0 = fmaf(wi0, x, fmaf(wh0, h, g0b));
            float g1 = fmaf(wi1, x, fmaf(wh1, h, g1b));
            float g2 = fmaf(wi2, x, fmaf(wh2, h, g2b));
            float g3 = fmaf(wi3, x, fmaf(wh3, h, g3b));
            float ig = sigmoid_fast(g0);
            float fg = sigmoid_fast(g1);
            float gg = tanh_fast(g2);
            float og = sigmoid_fast(g3);
            c = fmaf(fg, c, ig * gg);
            h = og * tanh_fast(c);
            flat[2 * s + t] = h;
        }
    }
}

// ---------------------------------------------------------------------------
// Head: y[g] = sigmoid( sum_k flat[g*10+k] * W3[k] + b3 )
// ---------------------------------------------------------------------------
__global__ __launch_bounds__(256)
void k_head(const float* __restrict__ flat,
            const float* __restrict__ W3,
            const float* __restrict__ b3,
            float* __restrict__ out) {
    const int g = blockIdx.x * blockDim.x + threadIdx.x;
    const int ngrp = 2 * SEQ / 10;  // 2048
    if (g < ngrp) {
        float sum = b3[0];
        #pragma unroll
        for (int k = 0; k < 10; ++k) sum = fmaf(flat[g * 10 + k], W3[k], sum);
        out[g] = sigmoid_fast(sum);
    }
}

extern "C" void kernel_launch(void* const* d_in, const int* in_sizes, int n_in,
                              void* d_out, int out_size, void* d_ws, size_t ws_size,
                              hipStream_t stream) {
    const float* x0    = (const float*)d_in[0];
    const float* x1    = (const float*)d_in[1];
    const float* W1    = (const float*)d_in[2];
    const float* b1    = (const float*)d_in[3];
    const float* W2    = (const float*)d_in[4];
    const float* b2    = (const float*)d_in[5];
    const float* Wih_f = (const float*)d_in[6];
    const float* Whh_f = (const float*)d_in[7];
    const float* bih_f = (const float*)d_in[8];
    const float* bhh_f = (const float*)d_in[9];
    const float* Wih_b = (const float*)d_in[10];
    const float* Whh_b = (const float*)d_in[11];
    const float* bih_b = (const float*)d_in[12];
    const float* bhh_b = (const float*)d_in[13];
    const float* W3    = (const float*)d_in[14];
    const float* b3    = (const float*)d_in[15];
    float* out = (float*)d_out;

    const size_t w1t_bytes  = (size_t)DIM * KDIM * 2;                 // 16.8 MB
    const size_t tail_bytes = (size_t)(NPART + 2) * SEQ * sizeof(float);
    dim3 gg(SEQ / MT, DIM / NT);   // 80 x 8

    if (ws_size >= w1t_bytes + tail_bytes) {
        __bf16* W1t  = (__bf16*)d_ws;
        float*  xpart = (float*)((char*)d_ws + w1t_bytes);
        float*  flat  = xpart + (size_t)NPART * SEQ;
        k_prep<<<dim3(DIM / 32, KDIM / 32), 256, 0, stream>>>(W1, W1t);
        k_gemm_tdm<<<gg, 256, 0, stream>>>(x0, x1, W1t, b1, W2, xpart);
        k_lstm<<<1, 64, 0, stream>>>(xpart, b2, Wih_f, Whh_f, bih_f, bhh_f,
                                     Wih_b, Whh_b, bih_b, bhh_b, flat);
        k_head<<<(2048 + 255) / 256, 256, 0, stream>>>(flat, W3, b3, out);
    } else {
        float* xpart = (float*)d_ws;
        float* flat  = xpart + (size_t)NPART * SEQ;
        k_gemm_direct<<<gg, 256, 0, stream>>>(x0, x1, W1, b1, W2, xpart);
        k_lstm<<<1, 64, 0, stream>>>(xpart, b2, Wih_f, Whh_f, bih_f, bhh_f,
                                     Wih_b, Whh_b, bih_b, bhh_b, flat);
        k_head<<<(2048 + 255) / 256, 256, 0, stream>>>(flat, W3, b3, out);
    }
}
